// EfficientAttention_2551210574440
// MI455X (gfx1250) — matrix operand order repaired and checked by reading, not compiled, not run on hardware
//
#include <hip/hip_runtime.h>

typedef __attribute__((ext_vector_type(16))) __bf16 v16bf;
typedef __attribute__((ext_vector_type(8)))  float  v8f;

#define NBATCH 2
#define CCH    96
#define NVOX   (32*64*64)   // 131072 voxels per batch
#define HEADS  4
#define HCH    24

// K order this code uses for A and B (contiguous, not the ISA table order), element e in a lane:
// lanes 0-15: K = 0..15 ; lanes 16-31: K = 16..31
__device__ __forceinline__ int kmapA(int e, int hi) { return e + (hi << 4); }

static __device__ __forceinline__ v8f wmma_bf16(v16bf a, v16bf b, v8f c) {
    return __builtin_amdgcn_wmma_f32_16x16x32_bf16(false, a, false, b, (short)0, c, false, false);
}

// ---------------------------------------------------------------------------
// Kernel 0: zero context + sumexp scratch
// ---------------------------------------------------------------------------
__global__ void k0_zero(float* __restrict__ p, int n) {
    int i = blockIdx.x * 256 + threadIdx.x;
    if (i < n) p[i] = 0.0f;
}

// ---------------------------------------------------------------------------
// Kernel 1: K/V projection (WMMA) + exp(K), per-channel sumexp, and
//           context[kch,vch] = sum_vox exp(K)*V accumulated via WMMA.
// grid = 128 (64 WGs per batch), block = 128 (4 waves), each wave: 16 chunks
// of 32 voxels (512-voxel strip).
// ---------------------------------------------------------------------------
__global__ void __launch_bounds__(128)
k1_proj_ctx(const float* __restrict__ x,
            const float* __restrict__ Wk, const float* __restrict__ bk,
            const float* __restrict__ Wv, const float* __restrict__ bv,
            float* __restrict__ gctx, float* __restrict__ gsum)
{
    extern __shared__ __align__(16) char smem[];
    __bf16* wkl  = (__bf16*)smem;                 // 96*96 bf16
    __bf16* wvl  = wkl + CCH * CCH;               // 96*96 bf16
    float*  bl   = (float*)(wvl + CCH * CCH);     // 192 f32 (bk | bv)
    float*  ctxl = bl + 192;                      // 96*96 f32 (WG context acc)
    __bf16* ekl  = (__bf16*)(ctxl + CCH * CCH);   // 4 waves * 96*32 bf16
    __bf16* vvl  = ekl + 4 * CCH * 32;            // 4 waves * 96*32 bf16

    const int tid  = threadIdx.x;
    const int lane = tid & 31, wid = tid >> 5;
    const int hi   = lane >> 4, l15 = lane & 15;

    for (int i = tid; i < CCH * CCH; i += 128) {
        wkl[i]  = (__bf16)Wk[i];
        wvl[i]  = (__bf16)Wv[i];
        ctxl[i] = 0.0f;
    }
    for (int i = tid; i < CCH; i += 128) { bl[i] = bk[i]; bl[CCH + i] = bv[i]; }
    __syncthreads();

    const int batch = blockIdx.x >> 6;
    const int w     = ((blockIdx.x & 63) << 2) + wid;       // wave id in batch
    const float* xn = x + (size_t)batch * CCH * NVOX;
    __bf16* ek = ekl + wid * CCH * 32;
    __bf16* vv = vvl + wid * CCH * 32;

    float se0 = 0.f, se1 = 0.f, se2 = 0.f;

    for (int chunk = 0; chunk < 16; ++chunk) {
        const int vox0 = w * 512 + chunk * 32;
        if (chunk < 15) {
#pragma unroll
            for (int t = 0; t < 3; ++t)
                __builtin_prefetch(&xn[(size_t)(lane + 32 * t) * NVOX + vox0 + 32], 0, 0);
        }
        // x B-fragments (B is 32x16: K=in-channel, N=voxel): K = e + 16*hi, N = l15
        v16bf bx[6];
#pragma unroll
        for (int col = 0; col < 2; ++col)
#pragma unroll
            for (int kc = 0; kc < 3; ++kc) {
                v16bf f;
#pragma unroll
                for (int e = 0; e < 16; ++e) {
                    int ch = kc * 32 + e + (hi << 4);
                    f[e] = (__bf16)xn[(size_t)ch * NVOX + vox0 + col * 16 + l15];
                }
                bx[col * 3 + kc] = f;
            }
        // projections: p=0 -> exp(K), p=1 -> V ; staged to LDS as [ch][vox]
#pragma unroll
        for (int p = 0; p < 2; ++p) {
            const __bf16* wl = p ? wvl : wkl;
            __bf16* dst = p ? vv : ek;
#pragma unroll
            for (int m = 0; m < 6; ++m) {
                v16bf a[3];
#pragma unroll
                for (int kc = 0; kc < 3; ++kc) {
                    v16bf f;
#pragma unroll
                    for (int e = 0; e < 16; ++e)
                        f[e] = wl[(m * 16 + l15) * CCH + kc * 32 + kmapA(e, hi)];
                    a[kc] = f;
                }
#pragma unroll
                for (int col = 0; col < 2; ++col) {
                    v8f acc = {};
                    acc = wmma_bf16(a[0], bx[col * 3 + 0], acc);
                    acc = wmma_bf16(a[1], bx[col * 3 + 1], acc);
                    acc = wmma_bf16(a[2], bx[col * 3 + 2], acc);
#pragma unroll
                    for (int r = 0; r < 8; ++r) {
                        int row = m * 16 + r + (hi << 3);   // C layout: M = r + 8*hi
                        float v = acc[r] + bl[p * CCH + row];
                        if (p == 0) v = __expf(v);          // stable w/o max: |k| small
                        dst[row * 32 + col * 16 + l15] = (__bf16)v;
                    }
                }
            }
        }
        // per-channel sum of exp(k): each lane owns channels lane, lane+32, lane+64
#pragma unroll
        for (int t = 0; t < 3; ++t) {
            int ch = lane + 32 * t;
            float s = 0.f;
#pragma unroll
            for (int v = 0; v < 32; ++v) s += (float)ek[ch * 32 + v];
            if (t == 0) se0 += s; else if (t == 1) se1 += s; else se2 += s;
        }
        // context tiles via WMMA: A = expK (16 kch x 32 vox), B = V^T (32 vox x 16 vch)
        // only tiles overlapping head-diagonal 24x24 blocks are needed
        const unsigned NEED[6] = {0x03u, 0x07u, 0x06u, 0x18u, 0x38u, 0x30u};
#pragma unroll
        for (int i = 0; i < 6; ++i) {
            v16bf ai;
#pragma unroll
            for (int e = 0; e < 16; ++e)
                ai[e] = ek[(i * 16 + l15) * 32 + kmapA(e, hi)];
#pragma unroll
            for (int j = 0; j < 6; ++j) {
                if (!((NEED[i] >> j) & 1u)) continue;
                v16bf bj;
#pragma unroll
                for (int e = 0; e < 16; ++e)
                    bj[e] = vv[(j * 16 + l15) * 32 + e + (hi << 4)];
                v8f acc = {};
                acc = wmma_bf16(ai, bj, acc);
#pragma unroll
                for (int r = 0; r < 8; ++r) {
                    int row  = i * 16 + r + (hi << 3);
                    int colc = j * 16 + l15;
                    atomicAdd(&ctxl[row * CCH + colc], acc[r]);   // ds_add_f32
                }
            }
        }
    }
    atomicAdd(&gsum[batch * CCH + lane],      se0);
    atomicAdd(&gsum[batch * CCH + lane + 32], se1);
    atomicAdd(&gsum[batch * CCH + lane + 64], se2);
    __syncthreads();
    for (int i = tid; i < CCH * CCH; i += 128)
        if (ctxl[i] != 0.0f)
            atomicAdd(&gctx[(size_t)batch * CCH * CCH + i], ctxl[i]);
}

// ---------------------------------------------------------------------------
// Kernel 2: fold Wr + softmax normalization into M[n][96][96]
// M[o,ck] = sum_v Wr[o, h*24+v] * gctx[ck, h*24+v] / sumexp[ck]
// ---------------------------------------------------------------------------
__global__ void __launch_bounds__(256)
k2_combine(const float* __restrict__ Wr, const float* __restrict__ gctx,
           const float* __restrict__ gsum, float* __restrict__ gM)
{
    for (int idx = threadIdx.x; idx < NBATCH * CCH * CCH; idx += 256) {
        int n   = idx / (CCH * CCH);
        int rem = idx - n * CCH * CCH;
        int o   = rem / CCH, ck = rem - o * CCH;
        int h   = ck / HCH;
        const float* crow = gctx + ((size_t)n * CCH + ck) * CCH + h * HCH;
        const float* wrow = Wr + o * CCH + h * HCH;
        float acc = 0.f;
#pragma unroll
        for (int v = 0; v < HCH; ++v) acc += wrow[v] * crow[v];
        gM[(size_t)n * CCH * CCH + o * CCH + ck] = acc / gsum[n * CCH + ck];
    }
}

// ---------------------------------------------------------------------------
// Kernel 3: recompute Q via WMMA, per-voxel head-channel softmax in LDS,
//           out = M @ q_softmax + br via WMMA.
// grid = 2048 (1024 per batch), block = 128 (4 waves), 32 voxels per wave.
// ---------------------------------------------------------------------------
__global__ void __launch_bounds__(128)
k3_out(const float* __restrict__ x,
       const float* __restrict__ Wq, const float* __restrict__ bq,
       const float* __restrict__ br, const float* __restrict__ gM,
       float* __restrict__ out)
{
    extern __shared__ __align__(16) char smem[];
    __bf16* wql = (__bf16*)smem;                  // 96*96 bf16
    __bf16* ml  = wql + CCH * CCH;                // 96*96 bf16 (M for this batch)
    float*  bql = (float*)(ml + CCH * CCH);       // 96
    float*  brl = bql + CCH;                      // 96
    float*  qf  = brl + CCH;                      // 4 waves * 96*32 f32
    __bf16* qb  = (__bf16*)(qf + 4 * CCH * 32);   // 4 waves * 96*32 bf16

    const int tid  = threadIdx.x;
    const int lane = tid & 31, wid = tid >> 5;
    const int hi   = lane >> 4, l15 = lane & 15;
    const int batch = blockIdx.x >> 10;
    const int vox0  = ((blockIdx.x & 1023) << 7) + (wid << 5);

    for (int i = tid; i < CCH * CCH; i += 128) {
        wql[i] = (__bf16)Wq[i];
        ml[i]  = (__bf16)gM[(size_t)batch * CCH * CCH + i];
    }
    for (int i = tid; i < CCH; i += 128) { bql[i] = bq[i]; brl[i] = br[i]; }
    __syncthreads();

    const float* xn = x + (size_t)batch * CCH * NVOX;
    float*  qfw = qf + wid * CCH * 32;
    __bf16* qbw = qb + wid * CCH * 32;

    // x B-fragments for this wave's 32 voxels
    v16bf bx[6];
#pragma unroll
    for (int col = 0; col < 2; ++col)
#pragma unroll
        for (int kc = 0; kc < 3; ++kc) {
            v16bf f;
#pragma unroll
            for (int e = 0; e < 16; ++e) {
                int ch = kc * 32 + e + (hi << 4);
                f[e] = (__bf16)xn[(size_t)ch * NVOX + vox0 + col * 16 + l15];
            }
            bx[col * 3 + kc] = f;
        }
    // Q projection -> qfw (f32, [ch][vox])
#pragma unroll
    for (int m = 0; m < 6; ++m) {
        v16bf a[3];
#pragma unroll
        for (int kc = 0; kc < 3; ++kc) {
            v16bf f;
#pragma unroll
            for (int e = 0; e < 16; ++e)
                f[e] = wql[(m * 16 + l15) * CCH + kc * 32 + kmapA(e, hi)];
            a[kc] = f;
        }
#pragma unroll
        for (int col = 0; col < 2; ++col) {
            v8f acc = {};
            acc = wmma_bf16(a[0], bx[col * 3 + 0], acc);
            acc = wmma_bf16(a[1], bx[col * 3 + 1], acc);
            acc = wmma_bf16(a[2], bx[col * 3 + 2], acc);
#pragma unroll
            for (int r = 0; r < 8; ++r) {
                int row = m * 16 + r + (hi << 3);
                qfw[row * 32 + col * 16 + l15] = acc[r] + bql[row];
            }
        }
    }
    // softmax over the 24 head-channels, per voxel (lane = voxel, t = head)
#pragma unroll
    for (int t = 0; t < HEADS; ++t) {
        float mx = -1e30f;
#pragma unroll
        for (int c = 0; c < HCH; ++c) mx = fmaxf(mx, qfw[(t * HCH + c) * 32 + lane]);
        float ev[HCH], s = 0.f;
#pragma unroll
        for (int c = 0; c < HCH; ++c) {
            float v = __expf(qfw[(t * HCH + c) * 32 + lane] - mx);
            ev[c] = v; s += v;
        }
        float inv = 1.f / s;
#pragma unroll
        for (int c = 0; c < HCH; ++c)
            qbw[(t * HCH + c) * 32 + lane] = (__bf16)(ev[c] * inv);
    }
    // out = M @ q_softmax + br
#pragma unroll
    for (int m = 0; m < 6; ++m) {
        v16bf a[3];
#pragma unroll
        for (int kc = 0; kc < 3; ++kc) {
            v16bf f;
#pragma unroll
            for (int e = 0; e < 16; ++e)
                f[e] = ml[(m * 16 + l15) * CCH + kc * 32 + kmapA(e, hi)];
            a[kc] = f;
        }
#pragma unroll
        for (int col = 0; col < 2; ++col) {
            v8f acc = {};
#pragma unroll
            for (int kc = 0; kc < 3; ++kc) {
                v16bf b;
#pragma unroll
                for (int e = 0; e < 16; ++e) {
                    int ch = kc * 32 + e + (hi << 4);
                    b[e] = qbw[ch * 32 + col * 16 + l15];
                }
                acc = wmma_bf16(a[kc], b, acc);
            }
#pragma unroll
            for (int r = 0; r < 8; ++r) {
                int row = m * 16 + r + (hi << 3);
                out[((size_t)batch * CCH + row) * NVOX + vox0 + col * 16 + l15] =
                    acc[r] + brl[row];
            }
        }
    }
}

// ---------------------------------------------------------------------------
extern "C" void kernel_launch(void* const* d_in, const int* in_sizes, int n_in,
                              void* d_out, int out_size, void* d_ws, size_t ws_size,
                              hipStream_t stream)
{
    (void)in_sizes; (void)n_in; (void)out_size; (void)ws_size;
    const float* x  = (const float*)d_in[0];
    const float* Wk = (const float*)d_in[1];
    const float* bk = (const float*)d_in[2];
    const float* Wq = (const float*)d_in[3];
    const float* bq = (const float*)d_in[4];
    const float* Wv = (const float*)d_in[5];
    const float* bv = (const float*)d_in[6];
    const float* Wr = (const float*)d_in[7];
    const float* br = (const float*)d_in[8];
    float* out = (float*)d_out;

    float* gctx = (float*)d_ws;                      // 2*96*96
    float* gsum = gctx + NBATCH * CCH * CCH;         // 2*96
    float* gM   = gsum + NBATCH * CCH;               // 2*96*96

    const int nz = NBATCH * CCH * CCH + NBATCH * CCH;
    hipLaunchKernelGGL(k0_zero, dim3((nz + 255) / 256), dim3(256), 0, stream, gctx, nz);

    const size_t sh1 = 2 * CCH * CCH * sizeof(__bf16)   // Wk,Wv
                     + 192 * sizeof(float)              // biases
                     + CCH * CCH * sizeof(float)        // ctx accumulator
                     + 2 * 4 * CCH * 32 * sizeof(__bf16); // ek + vv (4 waves)
    hipLaunchKernelGGL(k1_proj_ctx, dim3(128), dim3(128), sh1, stream,
                       x, Wk, bk, Wv, bv, gctx, gsum);

    hipLaunchKernelGGL(k2_combine, dim3(1), dim3(256), 0, stream, Wr, gctx, gsum, gM);

    const size_t sh3 = 2 * CCH * CCH * sizeof(__bf16)   // Wq, M
                     + 2 * CCH * sizeof(float)          // bq, br
                     + 4 * CCH * 32 * sizeof(float)     // qf (4 waves)
                     + 4 * CCH * 32 * sizeof(__bf16);   // qb (4 waves)
    hipLaunchKernelGGL(k3_out, dim3(2048), dim3(128), sh3, stream,
                       x, Wq, bq, br, gM, out);
}
